// Local_samplenet_77137612636425
// MI455X (gfx1250) — matrix operand order repaired and checked by reading, not compiled, not run on hardware
//
#include <hip/hip_runtime.h>
#include <math.h>

// ---------------------------------------------------------------------------
// Problem constants (match the JAX reference)
// ---------------------------------------------------------------------------
#define Bsz   64
#define Npts  8192
#define Pp    8
#define Ss    1024
#define Mm    8
#define Kk    7
#define BP    (Bsz * Pp)          // 512

typedef __attribute__((ext_vector_type(16))) __bf16 v16bf;
typedef __attribute__((ext_vector_type(8)))  float  v8f;

// ---------------------------------------------------------------------------
// WMMA fragment helpers (CDNA5 wave32 layouts, CDNA5 ISA §7.12.2)
//  A 16x32 bf16 : lane<16 -> M=lane, K = {0..7, 16..23}; lane>=16 -> K+8
//  B 32x16 bf16 : lane<16 -> N=lane, K = {0..7, 16..23}; lane>=16 -> K+8
//  C/D 16x16 f32: vgpr r -> M = r + (lane>=16 ? 8 : 0), N = lane&15
// ---------------------------------------------------------------------------
__device__ __forceinline__ v16bf load_a_lds(const __bf16* src, int r0, int kk, int ld) {
  int l  = threadIdx.x & 31;
  int m  = l & 15;
  int kb = (l >= 16) ? 8 : 0;
  const __bf16* row = src + (r0 + m) * ld + kk + kb;
  v16bf a;
#pragma unroll
  for (int i = 0; i < 8; ++i)  a[i] = row[i];
#pragma unroll
  for (int i = 8; i < 16; ++i) a[i] = row[i + 8];
  return a;
}

__device__ __forceinline__ v16bf load_a_global(const float* X, int r0, int kk, int ld) {
  int l  = threadIdx.x & 31;
  int m  = l & 15;
  int kb = (l >= 16) ? 8 : 0;
  const float* row = X + (size_t)(r0 + m) * ld + kk + kb;
  v16bf a;
#pragma unroll
  for (int i = 0; i < 8; ++i)  a[i] = (__bf16)row[i];
#pragma unroll
  for (int i = 8; i < 16; ++i) a[i] = (__bf16)row[i + 8];
  return a;
}

// Packed (fragment-major) B load: one unguarded 32-byte load per lane.
// Layout: (((tk * nTilesN + tn) * 32 + lane) * 16 + i)
__device__ __forceinline__ v16bf load_b_packed(const __bf16* Wp, int tk, int tn,
                                               int nTilesN) {
  const int l = threadIdx.x & 31;
  const __bf16* src = Wp + (((size_t)(tk * nTilesN + tn)) * 32 + l) * 16;
  return *(const v16bf*)src;
}

// ---------------------------------------------------------------------------
// 0) One-shot weight packer: f32 [CinReal,CoutReal] -> zero-padded bf16
//    fragment-major [CinPad/32][CoutPad/16][32 lanes][16 elems]. Branchless.
// ---------------------------------------------------------------------------
__global__ __launch_bounds__(256) void pack_w_kernel(const float* __restrict__ W,
                                                     __bf16* __restrict__ dst,
                                                     int CinReal, int CoutReal,
                                                     int CinPad, int CoutPad) {
  const int idx = blockIdx.x * 256 + threadIdx.x;
  const int total = CinPad * CoutPad;
  if (idx >= total) return;
  const int nTilesN = CoutPad >> 4;
  const int i    = idx & 15;
  const int l    = (idx >> 4) & 31;
  const int rest = idx >> 9;
  const int tn   = rest % nTilesN;
  const int tk   = rest / nTilesN;
  const int r = tk * 32 + ((l >= 16) ? 8 : 0) + i + ((i >= 8) ? 8 : 0);
  const int n = tn * 16 + (l & 15);
  const int rc = min(r, CinReal - 1);
  const int nc = min(n, CoutReal - 1);
  float v = W[(size_t)rc * CoutReal + nc];      // always in-bounds
  if (r >= CinReal || n >= CoutReal) v = 0.0f;  // v_cndmask, no branch
  dst[idx] = (__bf16)v;
}

// ---------------------------------------------------------------------------
// 1) Farthest point sampling: one block per batch, seeds coords out [B,P,3]
// ---------------------------------------------------------------------------
__global__ __launch_bounds__(256) void fps_kernel(const float* __restrict__ xyz,
                                                  float* __restrict__ seeds) {
  __shared__ float dist[Npts];
  __shared__ float rval[256];
  __shared__ int   ridx[256];
  __shared__ float cx, cy, cz;
  __shared__ int   farIdx;
  const int b = blockIdx.x;
  const int tid = threadIdx.x;
  const float* X = xyz + (size_t)b * 3 * Npts;

  for (int n = tid; n < Npts; n += 256) dist[n] = 1e10f;
  if (tid == 0) farIdx = 0;
  __syncthreads();

  for (int it = 0; it < Pp; ++it) {
    if (tid == 0) {
      int f = farIdx;
      cx = X[f]; cy = X[Npts + f]; cz = X[2 * Npts + f];
      float* s = seeds + ((size_t)b * Pp + it) * 3;
      s[0] = cx; s[1] = cy; s[2] = cz;
    }
    __syncthreads();
    float bv = -1.0f; int bi = 0x7fffffff;
    for (int n = tid; n < Npts; n += 256) {
      float dx = X[n] - cx, dy = X[Npts + n] - cy, dz = X[2 * Npts + n] - cz;
      float nd = fminf(dist[n], dx * dx + dy * dy + dz * dz);
      dist[n] = nd;
      if (nd > bv) { bv = nd; bi = n; }
    }
    rval[tid] = bv; ridx[tid] = bi;
    __syncthreads();
    for (int s2 = 128; s2 > 0; s2 >>= 1) {
      if (tid < s2) {
        float v2 = rval[tid + s2]; int i2 = ridx[tid + s2];
        if (v2 > rval[tid] || (v2 == rval[tid] && i2 < ridx[tid])) {
          rval[tid] = v2; ridx[tid] = i2;
        }
      }
      __syncthreads();
    }
    if (tid == 0) farIdx = ridx[0];
    __syncthreads();
  }
}

// ---------------------------------------------------------------------------
// 2) KNN (radix-select 1024 smallest of 8192) + patch normalization.
//    One block per (b,p). Writes patches [BP,S,3] (normed), mean[BP,3], norm[BP].
// ---------------------------------------------------------------------------
__global__ __launch_bounds__(256) void knn_patch_kernel(const float* __restrict__ xyz,
                                                        const float* __restrict__ seeds,
                                                        float* __restrict__ patches,
                                                        float* __restrict__ meanOut,
                                                        float* __restrict__ normOut) {
  __shared__ unsigned keys[Npts];          // 32 KB
  __shared__ unsigned hist[256];
  __shared__ unsigned exLess[256], exEq[256];
  __shared__ unsigned idxSel[Ss];
  __shared__ float    px[Ss], py[Ss], pz[Ss];
  __shared__ float    red[256];
  __shared__ unsigned sPrefix, sK, sNLess;
  __shared__ float    mx_, my_, mz_, nrm_;

  const int bp = blockIdx.x;
  const int b  = bp >> 3;
  const int tid = threadIdx.x;
  const float* X = xyz + (size_t)b * 3 * Npts;
  const float sx = seeds[bp * 3 + 0];
  const float sy = seeds[bp * 3 + 1];
  const float sz = seeds[bp * 3 + 2];

  for (int n = tid; n < Npts; n += 256) {
    float dx = X[n] - sx, dy = X[Npts + n] - sy, dz = X[2 * Npts + n] - sz;
    keys[n] = __float_as_uint(dx * dx + dy * dy + dz * dz);  // nonneg: uint order = float order
  }
  if (tid == 0) { sPrefix = 0u; sK = Ss; }
  __syncthreads();

  // 4-pass radix select of the k-th (k=1024) smallest key
  for (int pass = 0; pass < 4; ++pass) {
    const int shift = 24 - pass * 8;
    hist[tid] = 0u;
    __syncthreads();
    const unsigned prefix   = sPrefix;
    const unsigned maskHigh = (pass == 0) ? 0u : (0xFFFFFFFFu << (shift + 8));
    for (int n = tid; n < Npts; n += 256) {
      unsigned k = keys[n];
      if ((k & maskHigh) == prefix) atomicAdd(&hist[(k >> shift) & 0xFFu], 1u);
    }
    __syncthreads();
    if (tid == 0) {
      unsigned cum = 0, need = sK, bin = 0;
      for (bin = 0; bin < 256; ++bin) {
        if (cum + hist[bin] >= need) break;
        cum += hist[bin];
      }
      sPrefix = prefix | (bin << shift);
      sK = need - cum;
    }
    __syncthreads();
  }
  const unsigned T = sPrefix;   // exact 1024-th smallest key

  // deterministic compaction: strictly-less first, then ties in scan order
  unsigned nLess = 0, nEq = 0;
  for (int n = tid; n < Npts; n += 256) {
    unsigned k = keys[n];
    nLess += (k < T); nEq += (k == T);
  }
  exLess[tid] = nLess; exEq[tid] = nEq;
  __syncthreads();
  if (tid == 0) {
    unsigned aL = 0, aE = 0;
    for (int i = 0; i < 256; ++i) {
      unsigned tL = exLess[i], tE = exEq[i];
      exLess[i] = aL; exEq[i] = aE;
      aL += tL; aE += tE;
    }
    sNLess = aL;
  }
  __syncthreads();
  unsigned baseL = exLess[tid];
  unsigned baseE = sNLess + exEq[tid];
  for (int n = tid; n < Npts; n += 256) {
    unsigned k = keys[n];
    if (k < T) {
      idxSel[baseL++] = (unsigned)n;
    } else if (k == T) {
      unsigned slot = baseE++;
      if (slot < Ss) idxSel[slot] = (unsigned)n;
    }
  }
  __syncthreads();

  // gather patch coords
  for (int s = tid; s < Ss; s += 256) {
    unsigned n = idxSel[s];
    px[s] = X[n]; py[s] = X[Npts + n]; pz[s] = X[2 * Npts + n];
  }
  __syncthreads();

  // mean
  float sxa = 0, sya = 0, sza = 0;
  for (int s = tid; s < Ss; s += 256) { sxa += px[s]; sya += py[s]; sza += pz[s]; }
#define BLK_RED(var, op)                                                  \
  red[tid] = var; __syncthreads();                                        \
  for (int s2 = 128; s2 > 0; s2 >>= 1) {                                  \
    if (tid < s2) red[tid] = op(red[tid], red[tid + s2]);                 \
    __syncthreads();                                                      \
  }                                                                       \
  var = red[0]; __syncthreads();
#define ADDOP(a, b) ((a) + (b))
  BLK_RED(sxa, ADDOP)
  BLK_RED(sya, ADDOP)
  BLK_RED(sza, ADDOP)
  if (tid == 0) { mx_ = sxa / (float)Ss; my_ = sya / (float)Ss; mz_ = sza / (float)Ss; }
  __syncthreads();
  const float mx = mx_, my = my_, mz = mz_;

  // norm = max ||centered||
  float mmax = 0.0f;
  for (int s = tid; s < Ss; s += 256) {
    float dx = px[s] - mx, dy = py[s] - my, dz = pz[s] - mz;
    mmax = fmaxf(mmax, sqrtf(dx * dx + dy * dy + dz * dz));
  }
  BLK_RED(mmax, fmaxf)
  if (tid == 0) nrm_ = mmax;
  __syncthreads();
  const float inv = 1.0f / nrm_;

  for (int s = tid; s < Ss; s += 256) {
    float* o = patches + ((size_t)bp * Ss + s) * 3;
    o[0] = (px[s] - mx) * inv;
    o[1] = (py[s] - my) * inv;
    o[2] = (pz[s] - mz) * inv;
  }
  if (tid == 0) {
    meanOut[bp * 3 + 0] = mx; meanOut[bp * 3 + 1] = my; meanOut[bp * 3 + 2] = mz;
    normOut[bp] = nrm_;
  }
}

// ---------------------------------------------------------------------------
// 3) Fused per-point MLP (3->64->64->64->128->128, all ReLU) + max pool.
//    One block = 64 points of one patch; activations ping-pong in LDS (bf16).
//    Each wave computes TWO 16x16 N-tiles sharing one A fragment: two
//    independent v_wmma_f32_16x16x32_bf16 back-to-back fill the WMMA->VALU
//    hazard window and halve A-fragment traffic.
// ---------------------------------------------------------------------------
__device__ __forceinline__ void mlp_layer(const __bf16* src, __bf16* dst,
                                          const __bf16* Wp, const float* bias,
                                          int nTilesK, int Cout) {
  const int wave = threadIdx.x >> 5;
  const int nTilesN = Cout >> 4;
  const int nPair   = nTilesN >> 1;             // Cout is 64 or 128 -> even
  const int nTiles  = 4 * nPair;                // 4 row tiles (64 rows)
  const int l    = threadIdx.x & 31;
  const int roff = (l >= 16) ? 8 : 0;
  for (int t = wave; t < nTiles; t += 8) {
    const int r0  = (t & 3) << 4;
    const int tn0 = (t >> 2) << 1;
    v8f acc0 = {}, acc1 = {};
    for (int tk = 0; tk < nTilesK; ++tk) {
      v16bf a  = load_a_lds(src, r0, tk * 32, 128);
      v16bf b0 = load_b_packed(Wp, tk, tn0 + 0, nTilesN);
      v16bf b1 = load_b_packed(Wp, tk, tn0 + 1, nTilesN);
      acc0 = __builtin_amdgcn_wmma_f32_16x16x32_bf16(false, a, false, b0,
                                                     (short)0, acc0, false, false);
      acc1 = __builtin_amdgcn_wmma_f32_16x16x32_bf16(false, a, false, b1,
                                                     (short)0, acc1, false, false);
    }
    const int col0 = (tn0 << 4) + (l & 15);
    const float bv0 = bias[col0];
    const float bv1 = bias[col0 + 16];
#pragma unroll
    for (int r = 0; r < 8; ++r) {
      float v0 = fmaxf(acc0[r] + bv0, 0.0f);    // ReLU (all 5 MLP layers)
      float v1 = fmaxf(acc1[r] + bv1, 0.0f);
      dst[(r0 + r + roff) * 128 + col0]      = (__bf16)v0;
      dst[(r0 + r + roff) * 128 + col0 + 16] = (__bf16)v1;
    }
  }
  __syncthreads();
}

__global__ __launch_bounds__(256) void mlp_point_kernel(
    const float* __restrict__ patches,
    const __bf16* pw0, const float* mb0, const __bf16* pw1, const float* mb1,
    const __bf16* pw2, const float* mb2, const __bf16* pw3, const float* mb3,
    const __bf16* pw4, const float* mb4,
    unsigned* __restrict__ featBits) {
  __shared__ __bf16 actA[64 * 128];   // 16 KB
  __shared__ __bf16 actB[64 * 128];   // 16 KB
  const int blk   = blockIdx.x;
  const int bp    = blk >> 4;         // 16 chunks of 64 points per patch
  const int chunk = blk & 15;
  const size_t rowBase = (size_t)bp * Ss + (size_t)chunk * 64;
  const int tid = threadIdx.x;

  // stage layer-0 input: [64 x 32] with cols 3..31 zero-padded (branchless rows)
  if (tid < 64) {
    const float* pr = patches + (rowBase + tid) * 3;
    const float x = pr[0], y = pr[1], z = pr[2];
    __bf16* dr = actA + tid * 128;
    dr[0] = (__bf16)x; dr[1] = (__bf16)y; dr[2] = (__bf16)z;
#pragma unroll
    for (int c = 3; c < 32; ++c) dr[c] = (__bf16)0.0f;
  }
  __syncthreads();

  mlp_layer(actA, actB, pw0, mb0, 1,  64);   // 3(->32) -> 64
  mlp_layer(actB, actA, pw1, mb1, 2,  64);   // 64 -> 64
  mlp_layer(actA, actB, pw2, mb2, 2,  64);   // 64 -> 64
  mlp_layer(actB, actA, pw3, mb3, 2, 128);   // 64 -> 128
  mlp_layer(actA, actB, pw4, mb4, 4, 128);   // 128 -> 128

  // max pool over this chunk's 64 rows; merge across chunks with atomic max
  // (post-ReLU values are >= 0, so uint bit order == float order)
  if (tid < 128) {
    float mxv = 0.0f;
    for (int r = 0; r < 64; ++r) mxv = fmaxf(mxv, (float)actB[r * 128 + tid]);
    atomicMax(&featBits[(size_t)bp * 128 + tid], __float_as_uint(mxv));
  }
}

// ---------------------------------------------------------------------------
// 4) Generic WMMA GEMM for the patch decoder: Y = act(X @ W + b).
//    One wave per 16x32 output tile (two N-tiles, shared A fragment).
// ---------------------------------------------------------------------------
__global__ __launch_bounds__(32) void fc_gemm_kernel(const float* __restrict__ X,
                                                     const __bf16* __restrict__ Wp,
                                                     const float* __restrict__ bias,
                                                     float* __restrict__ Y,
                                                     int Cin, int Cout, int nTilesN,
                                                     int relu) {
  const int r0  = blockIdx.x * 16;
  const int tn0 = blockIdx.y << 1;
  v8f acc0 = {}, acc1 = {};
  for (int tk = 0; tk < (Cin >> 5); ++tk) {
    v16bf a  = load_a_global(X, r0, tk * 32, Cin);
    v16bf b0 = load_b_packed(Wp, tk, tn0 + 0, nTilesN);
    v16bf b1 = load_b_packed(Wp, tk, tn0 + 1, nTilesN);
    acc0 = __builtin_amdgcn_wmma_f32_16x16x32_bf16(false, a, false, b0,
                                                   (short)0, acc0, false, false);
    acc1 = __builtin_amdgcn_wmma_f32_16x16x32_bf16(false, a, false, b1,
                                                   (short)0, acc1, false, false);
  }
  const int l    = threadIdx.x & 31;
  const int roff = (l >= 16) ? 8 : 0;
  const int colA = (tn0 << 4) + (l & 15);
  const int colB = colA + 16;
  if (colA < Cout) {
    const float bv = bias[colA];
#pragma unroll
    for (int r = 0; r < 8; ++r) {
      float v = acc0[r] + bv;
      if (relu) v = fmaxf(v, 0.0f);
      Y[(size_t)(r0 + r + roff) * Cout + colA] = v;
    }
  }
  if (colB < Cout) {
    const float bv = bias[colB];
#pragma unroll
    for (int r = 0; r < 8; ++r) {
      float v = acc1[r] + bv;
      if (relu) v = fmaxf(v, 0.0f);
      Y[(size_t)(r0 + r + roff) * Cout + colB] = v;
    }
  }
}

// ---------------------------------------------------------------------------
// 5) Soft projection: per (b,p) block; K=7 nearest of S=1024, softmax(-d/sigma),
//    weighted sum, denormalize, write [B, P*M, 3].
// ---------------------------------------------------------------------------
__global__ __launch_bounds__(256) void softproj_kernel(const float* __restrict__ patches,
                                                       const float* __restrict__ gen,
                                                       const float* __restrict__ meanIn,
                                                       const float* __restrict__ normIn,
                                                       const float* __restrict__ tptr,
                                                       float* __restrict__ out) {
  __shared__ float px[Ss], py[Ss], pz[Ss];
  __shared__ float d[Ss];
  __shared__ float rv[256];
  __shared__ int   ri[256];
  __shared__ int   selI[Kk];
  __shared__ float selD[Kk];
  const int bp = blockIdx.x;
  const int b  = bp >> 3;
  const int p  = bp & 7;
  const int tid = threadIdx.x;

  for (int s = tid; s < Ss; s += 256) {
    const float* pp = patches + ((size_t)bp * Ss + s) * 3;
    px[s] = pp[0]; py[s] = pp[1]; pz[s] = pp[2];
  }
  const float t     = tptr[0];
  const float sigma = fmaxf(t * t, 1e-4f);
  const float m0 = meanIn[bp * 3 + 0];
  const float m1 = meanIn[bp * 3 + 1];
  const float m2 = meanIn[bp * 3 + 2];
  const float nrm = normIn[bp];
  __syncthreads();

  for (int m = 0; m < Mm; ++m) {
    const float gx = gen[(size_t)bp * (3 * Mm) + m * 3 + 0];
    const float gy = gen[(size_t)bp * (3 * Mm) + m * 3 + 1];
    const float gz = gen[(size_t)bp * (3 * Mm) + m * 3 + 2];
    for (int s = tid; s < Ss; s += 256) {
      float dx = px[s] - gx, dy = py[s] - gy, dz = pz[s] - gz;
      d[s] = dx * dx + dy * dy + dz * dz;
    }
    __syncthreads();
    for (int k = 0; k < Kk; ++k) {
      float bv = 3.4e38f; int bi = tid;
      for (int s = tid; s < Ss; s += 256) {
        float v = d[s];
        if (v < bv) { bv = v; bi = s; }
      }
      rv[tid] = bv; ri[tid] = bi;
      __syncthreads();
      for (int s2 = 128; s2 > 0; s2 >>= 1) {
        if (tid < s2) {
          float v2 = rv[tid + s2]; int i2 = ri[tid + s2];
          if (v2 < rv[tid] || (v2 == rv[tid] && i2 < ri[tid])) { rv[tid] = v2; ri[tid] = i2; }
        }
        __syncthreads();
      }
      if (tid == 0) { selI[k] = ri[0]; selD[k] = rv[0]; d[ri[0]] = 3.4e38f; }
      __syncthreads();
    }
    if (tid == 0) {
      const float mxl = -selD[0] / sigma;   // largest logit
      float wsum = 0, ox = 0, oy = 0, oz = 0;
      for (int k = 0; k < Kk; ++k) {
        float w = __expf(-selD[k] / sigma - mxl);
        int ii = selI[k];
        wsum += w;
        ox += w * px[ii]; oy += w * py[ii]; oz += w * pz[ii];
      }
      const float invw = 1.0f / wsum;
      float* o = out + ((size_t)b * (Pp * Mm) + (p * Mm + m)) * 3;
      o[0] = ox * invw * nrm + m0;
      o[1] = oy * invw * nrm + m1;
      o[2] = oz * invw * nrm + m2;
    }
    __syncthreads();
  }
}

__global__ void zero_kernel(float* __restrict__ ptr, int n) {
  int i = blockIdx.x * 256 + threadIdx.x;
  if (i < n) ptr[i] = 0.0f;
}

// ---------------------------------------------------------------------------
// Launcher
// ---------------------------------------------------------------------------
extern "C" void kernel_launch(void* const* d_in, const int* in_sizes, int n_in,
                              void* d_out, int out_size, void* d_ws, size_t ws_size,
                              hipStream_t stream) {
  (void)in_sizes; (void)n_in; (void)out_size; (void)ws_size;

  const float* xyz = (const float*)d_in[0];
  const float* mw0 = (const float*)d_in[1];  const float* mb0 = (const float*)d_in[2];
  const float* mw1 = (const float*)d_in[3];  const float* mb1 = (const float*)d_in[4];
  const float* mw2 = (const float*)d_in[5];  const float* mb2 = (const float*)d_in[6];
  const float* mw3 = (const float*)d_in[7];  const float* mb3 = (const float*)d_in[8];
  const float* mw4 = (const float*)d_in[9];  const float* mb4 = (const float*)d_in[10];
  const float* fw0 = (const float*)d_in[11]; const float* fb0 = (const float*)d_in[12];
  const float* fw1 = (const float*)d_in[13]; const float* fb1 = (const float*)d_in[14];
  const float* fw2 = (const float*)d_in[15]; const float* fb2 = (const float*)d_in[16];
  const float* fw3 = (const float*)d_in[17]; const float* fb3 = (const float*)d_in[18];
  const float* temperature = (const float*)d_in[19];

  // ---- workspace layout: f32 region then bf16 packed-weight region (~8.1 MB)
  float* ws      = (float*)d_ws;
  float* seeds   = ws;                               // B*P*3      = 1536
  float* patches = seeds   + (size_t)Bsz * Pp * 3;   // BP*S*3     = 1572864
  float* meanA   = patches + (size_t)BP * Ss * 3;    // BP*3       = 1536
  float* normA   = meanA   + (size_t)BP * 3;         // BP         = 512
  float* feat    = normA   + (size_t)BP;             // BP*128     = 65536
  float* fcA     = feat    + (size_t)BP * 128;       // BP*256     = 131072
  float* fcB     = fcA     + (size_t)BP * 256;       // BP*256     = 131072
  float* gen     = fcB     + (size_t)BP * 256;       // BP*24      = 12288
  __bf16* pw0 = (__bf16*)(gen + (size_t)BP * 24);    // 32*64   = 2048
  __bf16* pw1 = pw0 + 32 * 64;                       // 64*64   = 4096
  __bf16* pw2 = pw1 + 64 * 64;                       // 64*64   = 4096
  __bf16* pw3 = pw2 + 64 * 64;                       // 64*128  = 8192
  __bf16* pw4 = pw3 + 64 * 128;                      // 128*128 = 16384
  __bf16* pf0 = pw4 + 128 * 128;                     // 128*256 = 32768
  __bf16* pf1 = pf0 + 128 * 256;                     // 256*256 = 65536
  __bf16* pf2 = pf1 + 256 * 256;                     // 256*256 = 65536
  __bf16* pf3 = pf2 + 256 * 256;                     // 256*32  = 8192

  // ---- pack weights (bf16, zero-padded, fragment-major)
  pack_w_kernel<<<(32 * 64 + 255) / 256, 256, 0, stream>>>(mw0, pw0, 3, 64, 32, 64);
  pack_w_kernel<<<(64 * 64 + 255) / 256, 256, 0, stream>>>(mw1, pw1, 64, 64, 64, 64);
  pack_w_kernel<<<(64 * 64 + 255) / 256, 256, 0, stream>>>(mw2, pw2, 64, 64, 64, 64);
  pack_w_kernel<<<(64 * 128 + 255) / 256, 256, 0, stream>>>(mw3, pw3, 64, 128, 64, 128);
  pack_w_kernel<<<(128 * 128 + 255) / 256, 256, 0, stream>>>(mw4, pw4, 128, 128, 128, 128);
  pack_w_kernel<<<(128 * 256 + 255) / 256, 256, 0, stream>>>(fw0, pf0, 128, 256, 128, 256);
  pack_w_kernel<<<(256 * 256 + 255) / 256, 256, 0, stream>>>(fw1, pf1, 256, 256, 256, 256);
  pack_w_kernel<<<(256 * 256 + 255) / 256, 256, 0, stream>>>(fw2, pf2, 256, 256, 256, 256);
  pack_w_kernel<<<(256 * 32 + 255) / 256, 256, 0, stream>>>(fw3, pf3, 256, 24, 256, 32);

  // ---- pipeline
  fps_kernel<<<Bsz, 256, 0, stream>>>(xyz, seeds);
  knn_patch_kernel<<<BP, 256, 0, stream>>>(xyz, seeds, patches, meanA, normA);
  zero_kernel<<<(BP * 128 + 255) / 256, 256, 0, stream>>>(feat, BP * 128);
  mlp_point_kernel<<<BP * (Ss / 64), 256, 0, stream>>>(
      patches, pw0, mb0, pw1, mb1, pw2, mb2, pw3, mb3, pw4, mb4, (unsigned*)feat);

  // decoder: 512 rows -> 32 row tiles; grid.y = nTilesN/2 (paired N-tiles)
  fc_gemm_kernel<<<dim3(BP / 16, 8), 32, 0, stream>>>(feat, pf0, fb0, fcA, 128, 256, 16, 1);
  fc_gemm_kernel<<<dim3(BP / 16, 8), 32, 0, stream>>>(fcA,  pf1, fb1, fcB, 256, 256, 16, 1);
  fc_gemm_kernel<<<dim3(BP / 16, 8), 32, 0, stream>>>(fcB,  pf2, fb2, fcA, 256, 256, 16, 1);
  fc_gemm_kernel<<<dim3(BP / 16, 1), 32, 0, stream>>>(fcA,  pf3, fb3, gen, 256, 24, 2, 0);

  softproj_kernel<<<BP, 256, 0, stream>>>(patches, gen, meanA, normA, temperature, (float*)d_out);
}
